// OuterProductMean_88184268521492
// MI455X (gfx1250) — matrix operand order repaired and checked by reading, not compiled, not run on hardware
//
#include <hip/hip_runtime.h>

#define S_DIM 128
#define R_DIM 256
#define CM    256
#define H_DIM 32
#define CZ    128

typedef __attribute__((ext_vector_type(16))) __bf16 v16bf;
typedef __attribute__((ext_vector_type(8)))  __bf16 bf16x8;
typedef __attribute__((ext_vector_type(8)))  float  v8f;
typedef __attribute__((ext_vector_type(4)))  int    v4i;

// Pointer types for the async global->LDS builtin (param 0 is a 16-byte int
// vector pointer in the global AS per the round-2 diagnostic; LDS side mirrors
// it in AS3).
typedef __attribute__((address_space(1))) v4i* gv4i_ptr;
typedef __attribute__((address_space(3))) v4i* lv4i_ptr;

// ---- CDNA5 async global->LDS path (guarded; falls back to VGPR copy) ------
#if defined(__has_builtin)
#  if __has_builtin(__builtin_amdgcn_global_load_async_to_lds_b128)
#    define USE_ASYNC_COPY 1
#  endif
#  if __has_builtin(__builtin_amdgcn_s_wait_asynccnt)
#    define WAIT_ASYNC0() __builtin_amdgcn_s_wait_asynccnt(0)
#  endif
#endif
#ifndef USE_ASYNC_COPY
#  define USE_ASYNC_COPY 0
#endif
#ifndef WAIT_ASYNC0
#  define WAIT_ASYNC0() asm volatile("s_wait_asynccnt 0x0" ::: "memory")
#endif

// ---------------------------------------------------------------------------
// Fragment loaders: contiguous K order (lane half h holds K=16h..16h+15), same for A and B
// ---------------------------------------------------------------------------
// A-matrix 16x32 bf16: lanes 0-15 -> M=lane, K=0..15;
// lanes 16-31 -> M=lane-16, K=16..31.
__device__ __forceinline__ v16bf load_a_frag(const __bf16* base, int rstride, int lane) {
    const int m  = lane & 15;
    const int kl = (lane & 16) ? 16 : 0;
    const __bf16* p = base + m * rstride + kl;
    union { v16bf v; bf16x8 h[2]; } u;
    u.h[0] = *(const bf16x8*)(p);
    u.h[1] = *(const bf16x8*)(p + 8);
    return u.v;
}

// B-matrix 32x16 bf16 (KxN): lanes 0-15 -> N=lane, K=0..15; lanes 16-31 ->
// N=lane-16, K=16..31. Source must be column-major (N rows of contiguous K).
__device__ __forceinline__ v16bf load_b_frag(const __bf16* base, int cstride, int lane) {
    const int n  = lane & 15;
    const int kb = (lane & 16) ? 16 : 0;
    const __bf16* p = base + n * cstride + kb;
    union { v16bf v; bf16x8 h[2]; } u;
    u.h[0] = *(const bf16x8*)(p);
    u.h[1] = *(const bf16x8*)(p + 8);
    return u.v;
}

// ---------------------------------------------------------------------------
// Kernel 1: LayerNorm + projection + mask, store transposed bf16 workspace:
//   ws[r][h][s] = ((LN(m[s,r,:]) @ w)[h] + bias[h]) * mask[s,r]
// ---------------------------------------------------------------------------
__global__ __launch_bounds__(256) void lnproj_kernel(
    const float* __restrict__ m, const float* __restrict__ mask,
    const float* __restrict__ lnw, const float* __restrict__ lnb,
    const float* __restrict__ w, const float* __restrict__ bias,
    __bf16* __restrict__ ws)
{
    __shared__ float xrow[8][CM];
    const int tid  = threadIdx.x;
    const int lane = tid & 31;
    const int wv   = tid >> 5;
    const int row  = blockIdx.x * 8 + wv;      // row = s*R + r, 0..32767
    const int s    = row >> 8;
    const int r    = row & 255;

    const float4* xp = (const float4*)(m + (size_t)row * CM);
    const float4 x0 = xp[lane * 2 + 0];
    const float4 x1 = xp[lane * 2 + 1];
    float xs[8] = {x0.x, x0.y, x0.z, x0.w, x1.x, x1.y, x1.z, x1.w};

    float sum = xs[0] + xs[1] + xs[2] + xs[3] + xs[4] + xs[5] + xs[6] + xs[7];
#pragma unroll
    for (int off = 16; off > 0; off >>= 1) sum += __shfl_xor(sum, off, 32);
    const float mu = sum * (1.0f / (float)CM);

    float vs = 0.0f;
#pragma unroll
    for (int k = 0; k < 8; ++k) { const float d = xs[k] - mu; vs += d * d; }
#pragma unroll
    for (int off = 16; off > 0; off >>= 1) vs += __shfl_xor(vs, off, 32);
    const float inv = rsqrtf(vs * (1.0f / (float)CM) + 1e-5f);

    const float4* gp = (const float4*)lnw;
    const float4* bp = (const float4*)lnb;
    const float4 g0 = gp[lane * 2 + 0], g1 = gp[lane * 2 + 1];
    const float4 e0 = bp[lane * 2 + 0], e1 = bp[lane * 2 + 1];
    const float gs[8] = {g0.x, g0.y, g0.z, g0.w, g1.x, g1.y, g1.z, g1.w};
    const float es[8] = {e0.x, e0.y, e0.z, e0.w, e1.x, e1.y, e1.z, e1.w};
#pragma unroll
    for (int k = 0; k < 8; ++k)
        xrow[wv][lane * 8 + k] = (xs[k] - mu) * inv * gs[k] + es[k];
    __syncthreads();

    float acc = bias[lane];
    for (int k = 0; k < CM; ++k)
        acc += xrow[wv][k] * w[k * H_DIM + lane];
    acc *= mask[row];

    ws[((size_t)r * H_DIM + lane) * S_DIM + s] = (__bf16)acc;
}

// ---------------------------------------------------------------------------
// Kernel 2: transpose w_out [1024,128] f32 -> w_t [128,1024] bf16
// ---------------------------------------------------------------------------
__global__ __launch_bounds__(256) void wtrans_kernel(
    const float* __restrict__ w_out, __bf16* __restrict__ w_t)
{
    const int idx = blockIdx.x * 256 + threadIdx.x;   // idx = z*1024 + k
    const int z = idx >> 10;
    const int k = idx & 1023;
    w_t[idx] = (__bf16)w_out[k * CZ + z];
}

// ---------------------------------------------------------------------------
// Kernel 3: fused outer-product-mean + output projection.
// One workgroup per (16 i's) x (16 j's) tile pair. For each c in 0..31:
//   Stage 1: O_c[i,(j,e)] = sum_s a[s,i,c]*b[s,j,e]   (GEMM 16x512x128)
//   Stage 2: z[(i,j),z16] += O_c[(i,j),e] * W[c*32+e, z]  (K=32 per c)
// ---------------------------------------------------------------------------
__global__ __launch_bounds__(256) void fused_outer_kernel(
    const __bf16* __restrict__ a_ws,   // [R][H][S]
    const __bf16* __restrict__ b_ws,   // [R][H][S]
    const __bf16* __restrict__ w_t,    // [CZ][H*H]
    const float*  __restrict__ b_out,  // [CZ]
    const float*  __restrict__ mask1,  // [S][R]
    const float*  __restrict__ mask2,  // [S][R]
    float* __restrict__ out)           // [R][R][CZ]
{
    extern __shared__ char smem[];
    __bf16* aLDS = (__bf16*)smem;                       // 16*32*128  = 65536
    __bf16* bLDS = aLDS + 16 * H_DIM * S_DIM;           // 512*128    = 65536
    __bf16* oLDS = bLDS + 512 * S_DIM;                  // 2*16*16*32 = 16384
    float*  nLDS = (float*)(oLDS + 2 * 16 * 16 * H_DIM);// 256 floats

    const int tid  = threadIdx.x;
    const int lane = tid & 31;
    const int wv   = tid >> 5;
    const int i0   = (blockIdx.x >> 4) * 16;
    const int j0   = (blockIdx.x & 15) * 16;

    // Stage LDS: two contiguous 128KB tiles.
    {
        const int4* srcA = (const int4*)(a_ws + (size_t)i0 * H_DIM * S_DIM);
        const int4* srcB = (const int4*)(b_ws + (size_t)j0 * H_DIM * S_DIM);
        int4* dA = (int4*)aLDS;
        int4* dB = (int4*)bLDS;
        const int n16 = (16 * H_DIM * S_DIM) / 8;       // 8192 int4 each
#if USE_ASYNC_COPY
        // GLOBAL_LOAD_ASYNC_TO_LDS_B128: memory -> LDS without touching VGPRs,
        // tracked by ASYNCcnt (CDNA5 ISA).
        for (int k = tid; k < n16; k += 256) {
            __builtin_amdgcn_global_load_async_to_lds_b128(
                (gv4i_ptr)(srcA + k), (lv4i_ptr)(dA + k), 0, 0);
            __builtin_amdgcn_global_load_async_to_lds_b128(
                (gv4i_ptr)(srcB + k), (lv4i_ptr)(dB + k), 0, 0);
        }
        WAIT_ASYNC0();
#else
        for (int k = tid; k < n16; k += 256) { dA[k] = srcA[k]; dB[k] = srcB[k]; }
#endif
    }
    // Mask normalization: nLDS[i*16+j] = sum_s m1[s,i]*m2[s,j] + eps
    {
        const int i = tid >> 4, j = tid & 15;
        float ns = 0.0f;
        for (int ss = 0; ss < S_DIM; ++ss)
            ns += mask1[ss * R_DIM + i0 + i] * mask2[ss * R_DIM + j0 + j];
        nLDS[tid] = ns + 1e-3f;
    }
    __syncthreads();

    const v8f vzero = {0.f, 0.f, 0.f, 0.f, 0.f, 0.f, 0.f, 0.f};
    v8f acc[16];
#pragma unroll
    for (int i = 0; i < 16; ++i) acc[i] = vzero;

    for (int c = 0; c < H_DIM; ++c) {
        __bf16* oBuf = oLDS + (c & 1) * (16 * 16 * H_DIM);

        // ---- Stage 1: rows i (M=16), cols (j,e); this wave owns 4 N-tiles.
        v16bf afrag[4];
#pragma unroll
        for (int k = 0; k < 4; ++k)
            afrag[k] = load_a_frag(aLDS + c * S_DIM + k * 32, H_DIM * S_DIM, lane);

#pragma unroll
        for (int tt = 0; tt < 4; ++tt) {
            const int t = wv * 4 + tt;                  // N-tile index 0..31
            const __bf16* bbase = bLDS + t * 16 * S_DIM;
            v8f o = vzero;
            v16bf bf = load_b_frag(bbase, S_DIM, lane); // pipeline depth 1
#pragma unroll
            for (int k = 0; k < 4; ++k) {
                v16bf bn = bf;
                if (k < 3) bn = load_b_frag(bbase + (k + 1) * 32, S_DIM, lane);
                o = __builtin_amdgcn_wmma_f32_16x16x32_bf16(
                        false, afrag[k], false, bf, (short)0, o, false, false);
                bf = bn;
            }
            // Scatter D tile (M=i, N=e half-row of j) to oBuf[i][j][e] bf16.
            const int j  = t >> 1;
            const int eb = (t & 1) * 16 + (lane & 15);
            const int mb = (lane & 16) ? 8 : 0;
#pragma unroll
            for (int v = 0; v < 8; ++v)
                oBuf[((mb + v) * 16 + j) * H_DIM + eb] = (__bf16)o[v];
        }
        __syncthreads();

        // ---- Stage 2: each wave owns z-slice [wv*16, wv*16+16).
        if (c + 1 < H_DIM)
            __builtin_prefetch(w_t + (size_t)(wv * 16) * (H_DIM * H_DIM) + (c + 1) * H_DIM, 0, 1);
        const v16bf wfrag =
            load_b_frag(w_t + (size_t)(wv * 16) * (H_DIM * H_DIM) + c * H_DIM,
                        H_DIM * H_DIM, lane);
        v16bf afcur = load_a_frag(oBuf, H_DIM, lane);   // pipeline depth 1
#pragma unroll
        for (int i = 0; i < 16; ++i) {
            v16bf afnext = afcur;
            if (i < 15) afnext = load_a_frag(oBuf + (i + 1) * 16 * H_DIM, H_DIM, lane);
            acc[i] = __builtin_amdgcn_wmma_f32_16x16x32_bf16(
                         false, afcur, false, wfrag, (short)0, acc[i], false, false);
            afcur = afnext;
        }
    }

    // ---- Epilogue: bias, normalize, store. C tile (per i): M=j rows, N=z.
    const int zn = wv * 16 + (lane & 15);
    const float bz = b_out[zn];
    const int mb = (lane & 16) ? 8 : 0;
#pragma unroll
    for (int i = 0; i < 16; ++i) {
#pragma unroll
        for (int v = 0; v < 8; ++v) {
            const int j = mb + v;
            const float val = (acc[i][v] + bz) / nLDS[i * 16 + j];
            out[((size_t)(i0 + i) * R_DIM + (j0 + j)) * CZ + zn] = val;
        }
    }
}

// ---------------------------------------------------------------------------
extern "C" void kernel_launch(void* const* d_in, const int* in_sizes, int n_in,
                              void* d_out, int out_size, void* d_ws, size_t ws_size,
                              hipStream_t stream) {
    (void)in_sizes; (void)n_in; (void)out_size; (void)ws_size;
    const float* m1    = (const float*)d_in[0];
    const float* m2    = (const float*)d_in[1];
    const float* mask1 = (const float*)d_in[2];
    const float* mask2 = (const float*)d_in[3];
    const float* ln1w  = (const float*)d_in[4];
    const float* ln1b  = (const float*)d_in[5];
    const float* ln2w  = (const float*)d_in[6];
    const float* ln2b  = (const float*)d_in[7];
    const float* w1    = (const float*)d_in[8];
    const float* b1    = (const float*)d_in[9];
    const float* w2    = (const float*)d_in[10];
    const float* b2    = (const float*)d_in[11];
    const float* w_out = (const float*)d_in[12];
    const float* bout  = (const float*)d_in[13];

    __bf16* a_ws = (__bf16*)d_ws;                         // 1M bf16 = 2MB
    __bf16* b_ws = a_ws + (size_t)R_DIM * H_DIM * S_DIM;  // 1M bf16 = 2MB
    __bf16* w_t  = b_ws + (size_t)R_DIM * H_DIM * S_DIM;  // 128K bf16 = 256KB

    lnproj_kernel<<<4096, 256, 0, stream>>>(m1, mask1, ln1w, ln1b, w1, b1, a_ws);
    lnproj_kernel<<<4096, 256, 0, stream>>>(m2, mask2, ln2w, ln2b, w2, b2, b_ws);
    wtrans_kernel<<<512, 256, 0, stream>>>(w_out, w_t);

    const size_t lds_bytes =
        (size_t)(16 * H_DIM + 512) * S_DIM * sizeof(__bf16)
        + (size_t)(2 * 16 * 16 * H_DIM) * sizeof(__bf16)
        + 256 * sizeof(float);   // 295,936 B < 320KB WGP LDS
    fused_outer_kernel<<<256, 256, lds_bytes, stream>>>(
        a_ws, b_ws, w_t, bout, mask1, mask2, (float*)d_out);
}
